// MABSINK_38147899523730
// MI455X (gfx1250) — matrix operand order repaired and checked by reading, not compiled, not run on hardware
//
#include <hip/hip_runtime.h>

typedef __attribute__((ext_vector_type(16))) __bf16 v16bf;
typedef __attribute__((ext_vector_type(8)))  float  v8f;

#define BB 4
#define NQS 2048
#define NKS 2048
#define DD 1024
#define NH 16
#define DSH 64

union AF { v16bf v; uint4 q[2]; };

__device__ inline __bf16 f2bf(float f) {
  union { float f; unsigned u; } x; x.f = f;
  unsigned r = x.u + 0x7FFFu + ((x.u >> 16) & 1u);   // round-to-nearest-even
  union { unsigned short s; __bf16 b; } y; y.s = (unsigned short)(r >> 16);
  return y.b;
}

__device__ inline v8f wmma_bf16(v16bf a, v16bf b, v8f c) {
  // D = A(16x32 bf16) * B(32x16 bf16) + C(16x16 f32)
  return __builtin_amdgcn_wmma_f32_16x16x32_bf16(
      /*neg_a=*/false, a, /*neg_b=*/false, b,
      /*c_mod=*/(short)0, c, /*reuse_a=*/false, /*reuse_b=*/false);
}

// ---- A fragment (16x32, MxK) from bf16 row-major [M][ld].
// lane: m = lane&15, hi = lane>>4; elems 0..7 -> K = k0+hi*8+e; 8..15 -> K = k0+16+hi*8+(e-8)
__device__ inline v16bf load_a_bf16(const __bf16* __restrict__ src, int ld, int m0, int k0) {
  int lane = threadIdx.x & 31;
  int m = lane & 15, hi = lane >> 4;
  const __bf16* p = src + (size_t)(m0 + m) * ld + k0 + hi * 8;
  AF f;
  f.q[0] = *(const uint4*)(p);
  f.q[1] = *(const uint4*)(p + 16);
  return f.v;
}

// ---- A fragment from fp32 row-major (used once per wave in attention for Q)
__device__ inline v16bf load_a_f32(const float* __restrict__ src, int ld, int m0, int k0) {
  int lane = threadIdx.x & 31;
  int m = lane & 15, hi = lane >> 4;
  const float* p = src + (size_t)(m0 + m) * ld + k0 + hi * 8;
  float4 f0 = *(const float4*)(p);
  float4 f1 = *(const float4*)(p + 4);
  float4 f2 = *(const float4*)(p + 16);
  float4 f3 = *(const float4*)(p + 20);
  v16bf a;
  a[0]=f2bf(f0.x); a[1]=f2bf(f0.y); a[2]=f2bf(f0.z); a[3]=f2bf(f0.w);
  a[4]=f2bf(f1.x); a[5]=f2bf(f1.y); a[6]=f2bf(f1.z); a[7]=f2bf(f1.w);
  a[8]=f2bf(f2.x); a[9]=f2bf(f2.y); a[10]=f2bf(f2.z); a[11]=f2bf(f2.w);
  a[12]=f2bf(f3.x); a[13]=f2bf(f3.y); a[14]=f2bf(f3.z); a[15]=f2bf(f3.w);
  return a;
}

// ---- B fragment (32x16, KxN) for column-major B, i.e. src is B^T row-major [N][ld]
// lane: n = lane&15, hi = lane>>4; elems 0..7 -> K = k0+hi*8+e; 8..15 -> K = k0+16+hi*8+(e-8)
__device__ inline v16bf load_b_bf16(const __bf16* __restrict__ src, int ld, int n0, int k0) {
  int lane = threadIdx.x & 31;
  int n = lane & 15, hi = lane >> 4;
  const __bf16* p = src + (size_t)(n0 + n) * ld + k0 + hi * 8;
  AF f;
  f.q[0] = *(const uint4*)(p);
  f.q[1] = *(const uint4*)(p + 16);
  return f.v;
}

// ---------------------------------------------------------------- convert
__global__ __launch_bounds__(256) void cvt_kernel(const float* __restrict__ in,
                                                  __bf16* __restrict__ out, int n) {
  int idx = (blockIdx.x * 256 + threadIdx.x) * 4;
  if (idx >= n) return;
  float4 f = *(const float4*)(in + idx);
  union { __bf16 b[4]; uint2 u; } pk;
  pk.b[0] = f2bf(f.x); pk.b[1] = f2bf(f.y); pk.b[2] = f2bf(f.z); pk.b[3] = f2bf(f.w);
  *(uint2*)(out + idx) = pk.u;
}

// ---------------------------------------------------------------- GEMM
// C[M,N] = A[M,K](bf16) * W[N,K](bf16)^T + bias[N]
// Wave tile: 32(M) x 64(N) -> 2 A-frags x 4 B-frags = 8 WMMAs per 32-wide k-step.
// MODE 0: fp32 row-major out.  MODE 1: bf16 row-major out.
// MODE 2: bf16 transposed-per-batch out: out[(b*N+col)*rpb + r], b=row/rpb, r=row%rpb.
// MODE 3: fp32 out = resid + relu(x).
template<int MODE>
__global__ __launch_bounds__(256) void gemm_kernel(
    const __bf16* __restrict__ A, const __bf16* __restrict__ Wb,
    const float* __restrict__ bias, void* __restrict__ outp,
    const float* __restrict__ resid, int M, int N, int K, int rpb)
{
  int lane = threadIdx.x & 31;
  int wave = threadIdx.x >> 5;
  int m0 = (blockIdx.x * 8 + wave) * 32;
  int n0 = blockIdx.y * 64;

  v8f acc0[4] = {v8f{}, v8f{}, v8f{}, v8f{}};
  v8f acc1[4] = {v8f{}, v8f{}, v8f{}, v8f{}};
  for (int k0 = 0; k0 < K; k0 += 32) {
    v16bf a0 = load_a_bf16(A, K, m0,      k0);
    v16bf a1 = load_a_bf16(A, K, m0 + 16, k0);
    #pragma unroll
    for (int j = 0; j < 4; ++j) {
      v16bf b = load_b_bf16(Wb, K, n0 + j * 16, k0);
      acc0[j] = wmma_bf16(a0, b, acc0[j]);
      acc1[j] = wmma_bf16(a1, b, acc1[j]);
    }
  }

  int hi = lane >> 4, nl = lane & 15;
  #pragma unroll
  for (int mf = 0; mf < 2; ++mf) {
    #pragma unroll
    for (int j = 0; j < 4; ++j) {
      int col = n0 + j * 16 + nl;
      float bb = bias[col];
      #pragma unroll
      for (int v = 0; v < 8; ++v) {
        int row = m0 + mf * 16 + v + 8 * hi;
        float x = (mf ? acc1[j][v] : acc0[j][v]) + bb;
        if (MODE == 0) {
          ((float*)outp)[(size_t)row * N + col] = x;
        } else if (MODE == 1) {
          ((__bf16*)outp)[(size_t)row * N + col] = f2bf(x);
        } else if (MODE == 2) {
          int bidx = row / rpb, r = row % rpb;
          ((__bf16*)outp)[((size_t)bidx * N + col) * rpb + r] = f2bf(x);
        } else {
          x = fmaxf(x, 0.0f) + resid[(size_t)row * N + col];
          ((float*)outp)[(size_t)row * N + col] = x;
        }
      }
    }
  }
}

// ---------------------------------------------------------------- attention
// Per wave: 16 query rows of one (head, batch). Flash-style online softmax.
// scores = Qp*Kp^T/32 ; A = softmax/nq ; O = Qp + A*Vp
__global__ __launch_bounds__(256) void attn_kernel(
    const float* __restrict__ Qp, const __bf16* __restrict__ Kpb,
    const __bf16* __restrict__ Vtb, float* __restrict__ O)
{
  __shared__ __align__(16) __bf16 psm[8 * 16 * 32];   // per-wave 16x32 prob tile
  int lane = threadIdx.x & 31, wave = threadIdx.x >> 5;
  int hi = lane >> 4, nl = lane & 15;
  int h = blockIdx.y, b = blockIdx.z;
  int q0 = (blockIdx.x * 8 + wave) * 16;
  int hc = h * DSH;

  const float*  Qb_ = Qp  + (size_t)b * NQS * DD;
  const __bf16* Kb_ = Kpb + (size_t)b * NKS * DD;
  const __bf16* Vb_ = Vtb + (size_t)b * DD * NKS;
  __bf16* myp = psm + wave * (16 * 32);

  // Q fragments: features 0..31 and 32..63 of this head (reused over all keys)
  v16bf aq0 = load_a_f32(Qb_, DD, q0, hc);
  v16bf aq1 = load_a_f32(Qb_, DD, q0, hc + 32);

  v8f o[4] = {v8f{}, v8f{}, v8f{}, v8f{}};
  float m[8], l[8];
  #pragma unroll
  for (int v = 0; v < 8; ++v) { m[v] = -3.0e38f; l[v] = 0.0f; }
  const float scale = 0.03125f;   // 1/sqrt(1024)

  for (int kt = 0; kt < NKS; kt += 32) {
    // scores for 32 keys: two 16-wide tiles, each K-reduced over 64 features
    v8f s0 = {}, s1 = {};
    s0 = wmma_bf16(aq0, load_b_bf16(Kb_, DD, kt,      hc),      s0);
    s0 = wmma_bf16(aq1, load_b_bf16(Kb_, DD, kt,      hc + 32), s0);
    s1 = wmma_bf16(aq0, load_b_bf16(Kb_, DD, kt + 16, hc),      s1);
    s1 = wmma_bf16(aq1, load_b_bf16(Kb_, DD, kt + 16, hc + 32), s1);

    float p0[8], p1[8], alpha[8];
    #pragma unroll
    for (int v = 0; v < 8; ++v) {
      float x0 = s0[v] * scale, x1 = s1[v] * scale;
      float t = fmaxf(x0, x1);                       // row max over the 32 keys:
      t = fmaxf(t, __shfl_xor(t, 1, 32));            // xor<16 stays in the 16-lane
      t = fmaxf(t, __shfl_xor(t, 2, 32));            // group holding this row
      t = fmaxf(t, __shfl_xor(t, 4, 32));
      t = fmaxf(t, __shfl_xor(t, 8, 32));
      float mn = fmaxf(m[v], t);
      float al = __expf(m[v] - mn);
      p0[v] = __expf(x0 - mn);
      p1[v] = __expf(x1 - mn);
      float rs = p0[v] + p1[v];
      rs += __shfl_xor(rs, 1, 32);
      rs += __shfl_xor(rs, 2, 32);
      rs += __shfl_xor(rs, 4, 32);
      rs += __shfl_xor(rs, 8, 32);
      l[v] = l[v] * al + rs;
      m[v] = mn;
      alpha[v] = al;
    }
    #pragma unroll
    for (int j = 0; j < 4; ++j)
      #pragma unroll
      for (int v = 0; v < 8; ++v) o[j][v] = o[j][v] * alpha[v];

    // stage prob tile (16 rows x 32 keys) into this wave's LDS slice
    #pragma unroll
    for (int v = 0; v < 8; ++v) {
      int r = v + 8 * hi;
      myp[r * 32 + nl]      = f2bf(p0[v]);
      myp[r * 32 + 16 + nl] = f2bf(p1[v]);
    }
    asm volatile("s_wait_dscnt 0x0" ::: "memory");   // wave-local LDS ordering

    v16bf pa = load_a_bf16(myp, 32, 0, 0);           // probs as A fragment (K=32 keys)
    #pragma unroll
    for (int j = 0; j < 4; ++j) {
      v16bf bv = load_b_bf16(Vb_, NKS, hc + j * 16, kt);  // V^T: [feature][key]
      o[j] = wmma_bf16(pa, bv, o[j]);
    }
  }

  const float invq = 1.0f / (float)NQS;
  #pragma unroll
  for (int j = 0; j < 4; ++j) {
    #pragma unroll
    for (int v = 0; v < 8; ++v) {
      int row = q0 + v + 8 * hi;
      int col = hc + j * 16 + nl;
      size_t idx = (size_t)row * DD + col;
      O[(size_t)b * NQS * DD + idx] = Qb_[idx] + (o[j][v] / l[v]) * invq;
    }
  }
}

// ---------------------------------------------------------------- layernorm
// Optionally also emits a bf16 copy (GEMM A operand) alongside the fp32 result.
template<bool BOUT>
__global__ __launch_bounds__(256) void ln_kernel(const float* __restrict__ in,
                                                 const float* __restrict__ g,
                                                 const float* __restrict__ be,
                                                 float* __restrict__ out,
                                                 __bf16* __restrict__ out2)
{
  __shared__ float red[16];
  int row = blockIdx.x, tid = threadIdx.x;
  const float* rp = in + (size_t)row * DD;
  float4 v = ((const float4*)rp)[tid];                 // 256 threads * 4 = 1024
  float s  = v.x + v.y + v.z + v.w;
  float s2 = v.x * v.x + v.y * v.y + v.z * v.z + v.w * v.w;
  #pragma unroll
  for (int msk = 1; msk < 32; msk <<= 1) {
    s  += __shfl_xor(s,  msk, 32);
    s2 += __shfl_xor(s2, msk, 32);
  }
  int lane = tid & 31, wv = tid >> 5;
  if (lane == 0) { red[wv] = s; red[8 + wv] = s2; }
  __syncthreads();
  s = 0.0f; s2 = 0.0f;
  #pragma unroll
  for (int w = 0; w < 8; ++w) { s += red[w]; s2 += red[8 + w]; }
  float mu = s * (1.0f / DD);
  float var = s2 * (1.0f / DD) - mu * mu;
  float rs = rsqrtf(var + 1e-5f);
  float4 gv = ((const float4*)g)[tid];
  float4 bv = ((const float4*)be)[tid];
  float4 ov;
  ov.x = (v.x - mu) * rs * gv.x + bv.x;
  ov.y = (v.y - mu) * rs * gv.y + bv.y;
  ov.z = (v.z - mu) * rs * gv.z + bv.z;
  ov.w = (v.w - mu) * rs * gv.w + bv.w;
  ((float4*)(out + (size_t)row * DD))[tid] = ov;
  if (BOUT) {
    union { __bf16 b[4]; uint2 u; } pk;
    pk.b[0] = f2bf(ov.x); pk.b[1] = f2bf(ov.y);
    pk.b[2] = f2bf(ov.z); pk.b[3] = f2bf(ov.w);
    ((uint2*)(out2 + (size_t)row * DD))[tid] = pk.u;
  }
}

// ---------------------------------------------------------------- launch
extern "C" void kernel_launch(void* const* d_in, const int* in_sizes, int n_in,
                              void* d_out, int out_size, void* d_ws, size_t ws_size,
                              hipStream_t stream)
{
  (void)in_sizes; (void)n_in; (void)out_size; (void)ws_size;
  const float* Q  = (const float*)d_in[0];
  const float* K  = (const float*)d_in[1];
  const float* Wq = (const float*)d_in[2];
  const float* bq = (const float*)d_in[3];
  const float* Wk = (const float*)d_in[4];
  const float* bk = (const float*)d_in[5];
  const float* Wv = (const float*)d_in[6];
  const float* bv = (const float*)d_in[7];
  const float* Wo = (const float*)d_in[8];
  const float* bo = (const float*)d_in[9];
  const float* g0 = (const float*)d_in[10];
  const float* b0 = (const float*)d_in[11];
  const float* g1 = (const float*)d_in[12];
  const float* b1 = (const float*)d_in[13];

  char* ws = (char*)d_ws;
  size_t off = 0;
  auto alloc = [&](size_t bytes) -> void* {
    void* p = ws + off;
    off += (bytes + 255) & ~(size_t)255;
    return p;
  };
  const size_t NE = (size_t)BB * NQS * DD;     // 8M elements
  __bf16* Wqb = (__bf16*)alloc((size_t)DD * DD * 2);
  __bf16* Wkb = (__bf16*)alloc((size_t)DD * DD * 2);
  __bf16* Wvb = (__bf16*)alloc((size_t)DD * DD * 2);
  __bf16* Wob = (__bf16*)alloc((size_t)DD * DD * 2);
  __bf16* Qib = (__bf16*)alloc(NE * 2);        // bf16 Q input; reused later as LN0 bf16 out
  __bf16* Kib = (__bf16*)alloc(NE * 2);        // bf16 K input
  float*  Qp  = (float*) alloc(NE * 4);        // fp32 projected Q (residual + attn A op)
  __bf16* Kpb = (__bf16*)alloc(NE * 2);        // bf16 projected K
  __bf16* Vtb = (__bf16*)alloc(NE * 2);        // bf16 projected V, transposed per batch
  float*  O   = (float*) alloc(NE * 4);        // attention out, then LN0 in place
  float*  O2  = Qp;                            // reuse: Qp dead after attention
  __bf16* Ob  = Qib;                           // reuse: Qib dead after projections

  const int celem = DD * DD;
  cvt_kernel<<<celem / 4 / 256, 256, 0, stream>>>(Wq, Wqb, celem);
  cvt_kernel<<<celem / 4 / 256, 256, 0, stream>>>(Wk, Wkb, celem);
  cvt_kernel<<<celem / 4 / 256, 256, 0, stream>>>(Wv, Wvb, celem);
  cvt_kernel<<<celem / 4 / 256, 256, 0, stream>>>(Wo, Wob, celem);
  cvt_kernel<<<(int)(NE / 4 / 256), 256, 0, stream>>>(Q, Qib, (int)NE);
  cvt_kernel<<<(int)(NE / 4 / 256), 256, 0, stream>>>(K, Kib, (int)NE);

  dim3 gb((BB * NQS) / (32 * 8), DD / 64);     // (32, 16), 8 waves/block, 32x64/wave
  gemm_kernel<0><<<gb, 256, 0, stream>>>(Qib, Wqb, bq, Qp,  nullptr, BB * NQS, DD, DD, 0);
  gemm_kernel<1><<<gb, 256, 0, stream>>>(Kib, Wkb, bk, Kpb, nullptr, BB * NKS, DD, DD, 0);
  gemm_kernel<2><<<gb, 256, 0, stream>>>(Kib, Wvb, bv, Vtb, nullptr, BB * NKS, DD, DD, NKS);

  dim3 ga(NQS / (16 * 8), NH, BB);             // (16, 16, 4)
  attn_kernel<<<ga, 256, 0, stream>>>(Qp, Kpb, Vtb, O);

  ln_kernel<true ><<<BB * NQS, 256, 0, stream>>>(O, g0, b0, O, Ob);
  gemm_kernel<3><<<gb, 256, 0, stream>>>(Ob, Wob, bo, O2, O, BB * NQS, DD, DD, 0);
  ln_kernel<false><<<BB * NQS, 256, 0, stream>>>(O2, g1, b1, (float*)d_out, nullptr);
}
